// GraphNetwork_48799418417435
// MI455X (gfx1250) — matrix operand order repaired and checked by reading, not compiled, not run on hardware
//
#include <hip/hip_runtime.h>
#include <hip/hip_bf16.h>

#define DEVINL __device__ __forceinline__

typedef __attribute__((ext_vector_type(16))) _Float16 v16h;
typedef __attribute__((ext_vector_type(8)))  _Float16 v8h;
typedef __attribute__((ext_vector_type(8)))  float    v8f;

constexpr int N_NODES = 50000;
constexpr int N_EDGES = 1600000;
constexpr int NB      = 16;
constexpr int D       = 32;   // feature dim
constexpr int H       = 64;   // MLP hidden
constexpr float LN_EPS = 1e-5f;

// ------------------------------------------------------------------
// utility kernels
// ------------------------------------------------------------------
__global__ void k_zero(float* __restrict__ p, int n) {
  int i = blockIdx.x * blockDim.x + threadIdx.x;
  if (i < n) p[i] = 0.0f;
}

// Pack W[K x NCOLS] (f32, row major) into WMMA B-operand order (f16):
// out[((kt*NT + nt)*32 + lane)*16 + e] = W[(kt*32 + 16*(lane>>4) + e)*NCOLS + nt*16 + (lane&15)]
__global__ void k_pack_w(const float* __restrict__ W, _Float16* __restrict__ out,
                         int KT, int NT, int NCOLS) {
  int t = blockIdx.x * blockDim.x + threadIdx.x;
  if (t >= KT * NT * 32) return;
  int lane = t & 31;
  int tile = t >> 5;
  int nt = tile % NT;
  int kt = tile / NT;
  int n = (lane & 15) + nt * 16;
  int kbase = kt * 32 + ((lane >> 4) << 4);
  _Float16* o = out + ((size_t)t << 4);
#pragma unroll
  for (int e = 0; e < 16; ++e) o[e] = (_Float16)W[(size_t)(kbase + e) * NCOLS + n];
}

// edges -> nodes scatter:  recv[col] += e, send[row] += e   (fp32 HW atomics)
__global__ void k_scatter(const float4* __restrict__ ea4, const int* __restrict__ ei,
                          float* __restrict__ recv, float* __restrict__ send) {
  int idx = blockIdx.x * blockDim.x + threadIdx.x;
  if (idx >= N_EDGES * 8) return;            // 8 float4 per 32-feature edge row
  int e = idx >> 3;
  int q = (idx & 7) << 2;
  float4 v = ea4[idx];
  int r = ei[e];
  int c = ei[N_EDGES + e];
  float* pr = recv + (size_t)c * D + q;
  unsafeAtomicAdd(pr + 0, v.x);
  unsafeAtomicAdd(pr + 1, v.y);
  unsafeAtomicAdd(pr + 2, v.z);
  unsafeAtomicAdd(pr + 3, v.w);
  float* ps = send + (size_t)r * D + q;
  unsafeAtomicAdd(ps + 0, v.x);
  unsafeAtomicAdd(ps + 1, v.y);
  unsafeAtomicAdd(ps + 2, v.z);
  unsafeAtomicAdd(ps + 3, v.w);
}

// ------------------------------------------------------------------
// WMMA helpers
// ------------------------------------------------------------------
DEVINL v16h cat8(v8h lo, v8h hi) {
  return __builtin_shufflevector(lo, hi, 0, 1, 2, 3, 4, 5, 6, 7,
                                 8, 9, 10, 11, 12, 13, 14, 15);
}

// packed B operand: 32 contiguous bytes per lane -> 2x ds_load_b128
DEVINL v16h load_Bpacked(const _Float16* Wp, int tile, int lane) {
  const _Float16* p = Wp + (((size_t)(tile * 32 + lane)) << 4);
  return cat8(*(const v8h*)p, *(const v8h*)(p + 8));
}

// One 16-row MLP tile: [16 x 32*KT] @ W1[32*KT x 64] ->ReLU-> @ W2[64 x 32]
// ->ReLU-> LayerNorm.  srcs[kt] points at this lane's row features (32 f32).
// Output: o0[r] = row (hi8+r), col (lane&15); o1[r] = col (lane&15)+16.
template<int KT>
DEVINL void mlp_tile(const float* const (&srcs)[KT],
                     const _Float16* __restrict__ W1p,   // packed, KT*4 tiles
                     const _Float16* __restrict__ W2p,   // packed, 2*2 tiles
                     const float* __restrict__ b1, const float* __restrict__ b2,
                     const float* __restrict__ gam, const float* __restrict__ bet,
                     _Float16* hid /* per-wave 16*H staging */, int lane,
                     float (&o0)[8], float (&o1)[8]) {
  const int n   = lane & 15;
  const int hi8 = (lane >> 4) << 3;

  // ---- layer 1: [16 x 32KT] x [32KT x 64] via 16x16x32 f16 WMMA ----
  v8f acc[4];
#pragma unroll
  for (int i = 0; i < 4; ++i) { v8f z = {}; acc[i] = z; }
#pragma unroll
  for (int kt = 0; kt < KT; ++kt) {
    // A-layout per lane: k = 16*(lane>>4) + e -> one run of 16 f32
    v16h a;
#pragma unroll
    for (int e = 0; e < 16; ++e) {
      int k = (hi8 << 1) + e;
      a[e] = (_Float16)srcs[kt][k];
    }
#pragma unroll
    for (int nt = 0; nt < 4; ++nt) {
      acc[nt] = __builtin_amdgcn_wmma_f32_16x16x32_f16(
          false, a, false, load_Bpacked(W1p, kt * 4 + nt, lane),
          (short)0, acc[nt], false, false);
    }
  }

  // bias + ReLU -> f16 hidden in LDS (row-major; scatter stores, vector loads)
#pragma unroll
  for (int nt = 0; nt < 4; ++nt) {
    float bia = b1[nt * 16 + n];
#pragma unroll
    for (int r = 0; r < 8; ++r) {
      float h = fmaxf(acc[nt][r] + bia, 0.0f);
      hid[(hi8 + r) * H + nt * 16 + n] = (_Float16)h;
    }
  }
  __builtin_amdgcn_wave_barrier();
  asm volatile("s_wait_dscnt 0" ::: "memory");   // wave-local LDS RAW
  __builtin_amdgcn_wave_barrier();

  // ---- layer 2: [16 x 64] x [64 x 32] ----
  v8f d2[2];
#pragma unroll
  for (int i = 0; i < 2; ++i) { v8f z = {}; d2[i] = z; }
  const int m = lane & 15;
#pragma unroll
  for (int kt = 0; kt < 2; ++kt) {
    // A-layout gather = one contiguous 32B run -> 2x ds_load_b128
    const _Float16* hp = hid + m * H + kt * 32 + (hi8 << 1);
    v16h a = cat8(*(const v8h*)hp, *(const v8h*)(hp + 8));
#pragma unroll
    for (int nt = 0; nt < 2; ++nt) {
      d2[nt] = __builtin_amdgcn_wmma_f32_16x16x32_f16(
          false, a, false, load_Bpacked(W2p, kt * 2 + nt, lane),
          (short)0, d2[nt], false, false);
    }
  }

  // bias + ReLU + LayerNorm(32) — row spread over a 16-lane half, 2 cols/lane
  float bb0 = b2[n], bb1 = b2[n + 16];
  float g0 = gam[n], g1 = gam[n + 16];
  float be0 = bet[n], be1 = bet[n + 16];
#pragma unroll
  for (int r = 0; r < 8; ++r) {
    float v0 = fmaxf(d2[0][r] + bb0, 0.0f);
    float v1 = fmaxf(d2[1][r] + bb1, 0.0f);
    float s = v0 + v1;
    float q = v0 * v0 + v1 * v1;
#pragma unroll
    for (int mk = 1; mk < 16; mk <<= 1) {
      s += __shfl_xor(s, mk, 32);
      q += __shfl_xor(q, mk, 32);
    }
    float mu   = s * (1.0f / 32.0f);
    float var  = q * (1.0f / 32.0f) - mu * mu;
    float rstd = rsqrtf(var + LN_EPS);
    o0[r] = g0 * (v0 - mu) * rstd + be0;
    o1[r] = g1 * (v1 - mu) * rstd + be1;
  }
}

// ------------------------------------------------------------------
// NodeModel: x_new = MLP([x, recv, send, u[batch]]), fused n2g scatter
// ------------------------------------------------------------------
__global__ void __launch_bounds__(128)
k_node_mlp(const float* __restrict__ x, const float* __restrict__ recv,
           const float* __restrict__ send, const float* __restrict__ u,
           const int* __restrict__ batch,
           const _Float16* __restrict__ W1p, const _Float16* __restrict__ W2p,
           const float* __restrict__ b1, const float* __restrict__ b2,
           const float* __restrict__ gam, const float* __restrict__ bet,
           float* __restrict__ xnew, float* __restrict__ n2g) {
  __shared__ _Float16 sW1[128 * H];        // 4*4 packed tiles
  __shared__ _Float16 sW2[H * D];          // 2*2 packed tiles
  __shared__ _Float16 sHid[4][16 * H];
  for (int i = threadIdx.x; i < 128 * H; i += blockDim.x) sW1[i] = W1p[i];
  for (int i = threadIdx.x; i < H * D; i += blockDim.x) sW2[i] = W2p[i];
  __syncthreads();

  const int lane = threadIdx.x & 31;
  const int wid  = threadIdx.x >> 5;
  const int tile = blockIdx.x * 4 + wid;
  if (tile >= N_NODES / 16) return;          // wave-uniform guard

  const int node = tile * 16 + (lane & 15);
  const float* srcs[4] = { x + (size_t)node * D, recv + (size_t)node * D,
                           send + (size_t)node * D, u + (size_t)batch[node] * D };
  float o0[8], o1[8];
  mlp_tile<4>(srcs, sW1, sW2, b1, b2, gam, bet, sHid[wid], lane, o0, o1);

  const int n = lane & 15;
  const int hi8 = (lane >> 4) << 3;
#pragma unroll
  for (int r = 0; r < 8; ++r) {
    int nd = tile * 16 + hi8 + r;
    xnew[(size_t)nd * D + n]      = o0[r];
    xnew[(size_t)nd * D + n + 16] = o1[r];
    int gb = batch[nd];
    unsafeAtomicAdd(&n2g[gb * D + n],      o0[r]);
    unsafeAtomicAdd(&n2g[gb * D + n + 16], o1[r]);
  }
}

// ------------------------------------------------------------------
// EdgeModel: e_new = MLP([e, x_new[row], x_new[col], u[batch[row]]]), fused e2g
// ------------------------------------------------------------------
__global__ void __launch_bounds__(128)
k_edge_mlp(const float* __restrict__ ea, const int* __restrict__ ei,
           const int* __restrict__ batch, const float* __restrict__ u,
           const float* __restrict__ xnew,
           const _Float16* __restrict__ W1p, const _Float16* __restrict__ W2p,
           const float* __restrict__ b1, const float* __restrict__ b2,
           const float* __restrict__ gam, const float* __restrict__ bet,
           float* __restrict__ enew, float* __restrict__ e2g) {
  __shared__ _Float16 sW1[128 * H];
  __shared__ _Float16 sW2[H * D];
  __shared__ _Float16 sHid[4][16 * H];
  for (int i = threadIdx.x; i < 128 * H; i += blockDim.x) sW1[i] = W1p[i];
  for (int i = threadIdx.x; i < H * D; i += blockDim.x) sW2[i] = W2p[i];
  __syncthreads();

  const int lane = threadIdx.x & 31;
  const int wid  = threadIdx.x >> 5;
  const int tile = blockIdx.x * 4 + wid;     // E/16 divisible by 4 -> no guard

  const int e0 = tile * 16 + (lane & 15);
  const int r0 = ei[e0];
  const int c0 = ei[N_EDGES + e0];
  const float* srcs[4] = { ea + (size_t)e0 * D, xnew + (size_t)r0 * D,
                           xnew + (size_t)c0 * D, u + (size_t)batch[r0] * D };
  float o0[8], o1[8];
  mlp_tile<4>(srcs, sW1, sW2, b1, b2, gam, bet, sHid[wid], lane, o0, o1);

  const int n = lane & 15;
  const int hi8 = (lane >> 4) << 3;
#pragma unroll
  for (int r = 0; r < 8; ++r) {
    int er = tile * 16 + hi8 + r;
    enew[(size_t)er * D + n]      = o0[r];
    enew[(size_t)er * D + n + 16] = o1[r];
    int gb = batch[ei[er]];
    unsafeAtomicAdd(&e2g[gb * D + n],      o0[r]);
    unsafeAtomicAdd(&e2g[gb * D + n + 16], o1[r]);
  }
}

// ------------------------------------------------------------------
// GlobalModel: u_new = MLP([u, n2g, e2g])  (B=16 -> one 16-row tile, one wave)
// ------------------------------------------------------------------
__global__ void k_glob_mlp(const float* __restrict__ u, const float* __restrict__ n2g,
                           const float* __restrict__ e2g,
                           const _Float16* __restrict__ W1p, const _Float16* __restrict__ W2p,
                           const float* __restrict__ b1, const float* __restrict__ b2,
                           const float* __restrict__ gam, const float* __restrict__ bet,
                           float* __restrict__ unew) {
  __shared__ _Float16 sHid[16 * H];
  const int lane = threadIdx.x & 31;
  const int b = lane & 15;
  const float* srcs[3] = { u + (size_t)b * D, n2g + (size_t)b * D, e2g + (size_t)b * D };
  float o0[8], o1[8];
  mlp_tile<3>(srcs, W1p, W2p, b1, b2, gam, bet, sHid, lane, o0, o1);
  const int n = lane & 15;
  const int hi8 = (lane >> 4) << 3;
#pragma unroll
  for (int r = 0; r < 8; ++r) {
    int g = hi8 + r;
    unew[g * D + n]      = o0[r];
    unew[g * D + n + 16] = o1[r];
  }
}

// ------------------------------------------------------------------
extern "C" void kernel_launch(void* const* d_in, const int* in_sizes, int n_in,
                              void* d_out, int out_size, void* d_ws, size_t ws_size,
                              hipStream_t stream) {
  const float* x     = (const float*)d_in[0];
  const float* ea    = (const float*)d_in[1];
  const int*   ei    = (const int*)d_in[2];
  const int*   batch = (const int*)d_in[3];
  const float* u     = (const float*)d_in[4];
  const float* nW1 = (const float*)d_in[5],  *nb1 = (const float*)d_in[6];
  const float* nW2 = (const float*)d_in[7],  *nb2 = (const float*)d_in[8];
  const float* ng  = (const float*)d_in[9],  *nbe = (const float*)d_in[10];
  const float* eW1 = (const float*)d_in[11], *eb1 = (const float*)d_in[12];
  const float* eW2 = (const float*)d_in[13], *eb2 = (const float*)d_in[14];
  const float* eg  = (const float*)d_in[15], *ebe = (const float*)d_in[16];
  const float* gW1 = (const float*)d_in[17], *gb1 = (const float*)d_in[18];
  const float* gW2 = (const float*)d_in[19], *gb2 = (const float*)d_in[20];
  const float* gg  = (const float*)d_in[21], *gbe = (const float*)d_in[22];

  float* xnew = (float*)d_out;
  float* enew = xnew + (size_t)N_NODES * D;
  float* unew = enew + (size_t)N_EDGES * D;

  // workspace layout
  float* recv = (float*)d_ws;                     // N*32
  float* send = recv + (size_t)N_NODES * D;       // N*32
  float* n2g  = send + (size_t)N_NODES * D;       // B*32
  float* e2g  = n2g + NB * D;                     // B*32
  _Float16* pW1n = (_Float16*)(e2g + NB * D);     // 4*4*512 = 8192
  _Float16* pW2n = pW1n + 128 * H;                // 2*2*512 = 2048
  _Float16* pW1e = pW2n + H * D;
  _Float16* pW2e = pW1e + 128 * H;
  _Float16* pW1g = pW2e + H * D;                  // 3*4*512 = 6144
  _Float16* pW2g = pW1g + 96 * H;

  const int nz = 2 * N_NODES * D + 2 * NB * D;
  k_zero<<<(nz + 255) / 256, 256, 0, stream>>>(recv, nz);

  // pack weights into WMMA B-operand order (f16)
  k_pack_w<<<(4 * 4 * 32 + 127) / 128, 128, 0, stream>>>(nW1, pW1n, 4, 4, H);
  k_pack_w<<<(2 * 2 * 32 + 127) / 128, 128, 0, stream>>>(nW2, pW2n, 2, 2, D);
  k_pack_w<<<(4 * 4 * 32 + 127) / 128, 128, 0, stream>>>(eW1, pW1e, 4, 4, H);
  k_pack_w<<<(2 * 2 * 32 + 127) / 128, 128, 0, stream>>>(eW2, pW2e, 2, 2, D);
  k_pack_w<<<(3 * 4 * 32 + 127) / 128, 128, 0, stream>>>(gW1, pW1g, 3, 4, H);
  k_pack_w<<<(2 * 2 * 32 + 127) / 128, 128, 0, stream>>>(gW2, pW2g, 2, 2, D);

  k_scatter<<<(N_EDGES * 8 + 255) / 256, 256, 0, stream>>>((const float4*)ea, ei, recv, send);

  k_node_mlp<<<(N_NODES / 16 + 3) / 4, 128, 0, stream>>>(
      x, recv, send, u, batch, pW1n, pW2n, nb1, nb2, ng, nbe, xnew, n2g);

  k_edge_mlp<<<(N_EDGES / 16) / 4, 128, 0, stream>>>(
      ea, ei, batch, u, xnew, pW1e, pW2e, eb1, eb2, eg, ebe, enew, e2g);

  k_glob_mlp<<<1, 32, 0, stream>>>(u, n2g, e2g, pW1g, pW2g, gb1, gb2, gg, gbe, unew);
}
